// drGAT_64939905516121
// MI455X (gfx1250) — compile-verified, matrix operand order checked by reading, not run on hardware
//
#include <hip/hip_runtime.h>
#include <hip/hip_bf16.h>

// ---------------------------------------------------------------------------
// drGAT on gfx1250: WMMA f16 GEMMs (async-to-LDS pipelined for aligned shapes)
// + atomic-based segment softmax/aggregation
// ---------------------------------------------------------------------------

typedef __attribute__((ext_vector_type(16))) _Float16 v16h;
typedef __attribute__((ext_vector_type(8)))  float    v8f;

union Frag16 { v16h v; uint4 q[2]; };
union H4 { _Float16 h[4]; uint2 u; };

#define NNODES 8000
#define HC     512      // heads*channels per layer output
#define CH     128      // channels per head
#define NHEADS 4

// ---- CDNA5 async copy global->LDS (ASYNCcnt), per ISA 15.18.3 / 08_async ---
__device__ __forceinline__ void async_ld_b128(unsigned lds_off, const float* g) {
    asm volatile("global_load_async_to_lds_b128 %0, %1, off"
                 :: "v"(lds_off), "v"((unsigned long long)(uintptr_t)g)
                 : "memory");
}
__device__ __forceinline__ void wait_async0() {
    asm volatile("s_wait_asynccnt 0x0" ::: "memory");
}

// fragment build per the documented 16-bit VGPR layout:
// lanes 0-15: elems 0..7 = K0..7, 8..15 = K16..23 ; lanes 16-31: +8
__device__ __forceinline__ void build_frag(Frag16& f, const _Float16* row, int kbase) {
    f.q[0] = *reinterpret_cast<const uint4*>(row + kbase);
    f.q[1] = *reinterpret_cast<const uint4*>(row + kbase + 16);
}

// ----------------------- WMMA GEMM (general shapes) -------------------------
// C[M,N] = A[M,K] * W[K,N] + bias[N]   (fp32 in/out, f16 WMMA, f32 accum)
// block tile 64x64, 256 threads = 8 waves, wave grid 4(M) x 2(N),
// each wave computes 16(M) x 32(N) via two v_wmma_f32_16x16x32_f16 per K-step.
__global__ __launch_bounds__(256) void gemm_f16_wmma(
    const float* __restrict__ A, const float* __restrict__ W,
    const float* __restrict__ bias, float* __restrict__ C,
    int M, int K, int N)
{
    __shared__ __align__(16) _Float16 As[64][32];   // [m][k]
    __shared__ __align__(16) _Float16 Bs[64][32];   // [n][k]  (W^T tile)

    const int t    = threadIdx.x;
    const int lane = t & 31;
    const int wave = t >> 5;
    const int wm   = wave >> 1;          // 0..3
    const int wn   = wave & 1;           // 0..1
    const int bm   = blockIdx.y * 64;
    const int bn   = blockIdx.x * 64;

    const bool fullM = (bm + 64 <= M);
    const bool fullN = (bn + 64 <= N);

    v8f acc0 = {}; v8f acc1 = {};

    for (int k0 = 0; k0 < K; k0 += 32) {
        const bool fullK = (k0 + 32 <= K);

        if (fullM && fullK) {
            // fast path: one b128 global load + one b64 LDS store per pass
            #pragma unroll
            for (int p = 0; p < 2; ++p) {
                int li = p * 256 + t;                 // [0,512)
                int r  = li >> 3;                     // 0..63
                int kk = (li & 7) * 4;                // 0..28
                float4 v = *reinterpret_cast<const float4*>(
                    A + (long)(bm + r) * K + k0 + kk);
                H4 h; h.h[0] = (_Float16)v.x; h.h[1] = (_Float16)v.y;
                      h.h[2] = (_Float16)v.z; h.h[3] = (_Float16)v.w;
                *reinterpret_cast<uint2*>(&As[r][kk]) = h.u;
            }
        } else {
            // branchless edge path: clamped addresses, unconditional loads
            float av[8];
            #pragma unroll
            for (int i = 0; i < 8; ++i) {
                int li = i * 256 + t;
                int gr = bm + (li >> 5), gk = k0 + (li & 31);
                int grc = gr < M ? gr : M - 1;
                int gkc = gk < K ? gk : K - 1;
                float v = A[(long)grc * K + gkc];
                av[i] = (gr < M && gk < K) ? v : 0.0f;
            }
            #pragma unroll
            for (int i = 0; i < 8; ++i) {
                int li = i * 256 + t;
                As[li >> 5][li & 31] = (_Float16)av[i];
            }
        }

        if (fullN && fullK) {
            // 4 coalesced b32 loads (consecutive lanes -> consecutive n),
            // packed into one b64 LDS store of 4 consecutive k
            #pragma unroll
            for (int p = 0; p < 2; ++p) {
                int li  = p * 256 + t;                // [0,512)
                int nn  = li & 63;
                int kk0 = (li >> 6) * 4;              // 0..28
                H4 h;
                #pragma unroll
                for (int j = 0; j < 4; ++j)
                    h.h[j] = (_Float16)W[(long)(k0 + kk0 + j) * N + bn + nn];
                *reinterpret_cast<uint2*>(&Bs[nn][kk0]) = h.u;
            }
        } else {
            float bv[8];
            #pragma unroll
            for (int i = 0; i < 8; ++i) {
                int li = i * 256 + t;
                int gk = k0 + (li >> 6), gn = bn + (li & 63);
                int gkc = gk < K ? gk : K - 1;
                int gnc = gn < N ? gn : N - 1;
                float v = W[(long)gkc * N + gnc];
                bv[i] = (gk < K && gn < N) ? v : 0.0f;
            }
            #pragma unroll
            for (int i = 0; i < 8; ++i) {
                int li = i * 256 + t;
                Bs[li & 63][li >> 6] = (_Float16)bv[i];
            }
        }

        // prefetch next K-tile into cache while this tile is consumed
        if (k0 + 32 < K) {
            int prA = bm + (t >> 3);
            if (prA < M)
                __builtin_prefetch(A + (long)prA * K + (k0 + 32) + (t & 7) * 4, 0, 3);
            int prK = k0 + 32 + (t >> 6) * 4;
            int prN = bn + (t & 63);
            if (prK < K && prN < N)
                __builtin_prefetch(W + (long)prK * N + prN, 0, 3);
        }
        __syncthreads();

        const int l15   = lane & 15;
        const int kbase = (lane >> 4) * 8;
        Frag16 fa, fb0, fb1;
        build_frag(fa,  &As[wm * 16 + l15][0], kbase);
        build_frag(fb0, &Bs[wn * 32 + l15][0], kbase);
        build_frag(fb1, &Bs[wn * 32 + 16 + l15][0], kbase);
        acc0 = __builtin_amdgcn_wmma_f32_16x16x32_f16(false, fa.v, false, fb0.v,
                                                      (short)0, acc0, false, false);
        acc1 = __builtin_amdgcn_wmma_f32_16x16x32_f16(false, fa.v, false, fb1.v,
                                                      (short)0, acc1, false, false);
        __syncthreads();
    }

    // C/D layout: VGPR j -> row = j + 8*(lane>=16), col = lane&15
    const int l15  = lane & 15;
    const int row0 = bm + wm * 16 + (lane >> 4) * 8;
    const int col0 = bn + wn * 32 + l15;
    const int col1 = col0 + 16;
    const float bc0 = (col0 < N) ? bias[col0] : 0.0f;
    const float bc1 = (col1 < N) ? bias[col1] : 0.0f;
    #pragma unroll
    for (int j = 0; j < 8; ++j) {
        int r = row0 + j;
        if (r < M) {
            if (col0 < N) C[(long)r * N + col0] = acc0[j] + bc0;
            if (col1 < N) C[(long)r * N + col1] = acc1[j] + bc1;
        }
    }
}

// ------------- WMMA GEMM, async-to-LDS double-buffered pipeline -------------
// Requires M%64==0, N%64==0, K%32==0 (layer GEMMs: 8000 x {256,512} x 512).
// Raw f32 tiles stream into LDS via ASYNCcnt one K-step ahead of the WMMAs.
__device__ __forceinline__ void issue_tiles_async(
    const float* __restrict__ A, const float* __restrict__ W, int K, int N,
    int bm, int bn, int k0, int t, float (*Af)[32], float (*Bf)[64])
{
    #pragma unroll
    for (int p = 0; p < 2; ++p) {           // A: 64x32 f32, 16B per lane-op
        int li = p * 256 + t;
        int r = li >> 3, kk = (li & 7) * 4;
        async_ld_b128((unsigned)(uintptr_t)&Af[r][kk],
                      A + (long)(bm + r) * K + k0 + kk);
    }
    #pragma unroll
    for (int p = 0; p < 2; ++p) {           // B: 32x64 f32, row-major copy
        int li = p * 256 + t;
        int kk = li >> 4, nn = (li & 15) * 4;
        async_ld_b128((unsigned)(uintptr_t)&Bf[kk][nn],
                      W + (long)(k0 + kk) * N + bn + nn);
    }
}

__global__ __launch_bounds__(256) void gemm_f16_wmma_async(
    const float* __restrict__ A, const float* __restrict__ W,
    const float* __restrict__ bias, float* __restrict__ C,
    int M, int K, int N)
{
    __shared__ __align__(16) float    Af32[2][64][32];  // 16KB raw staging
    __shared__ __align__(16) float    Bf32[2][32][64];  // 16KB raw staging
    __shared__ __align__(16) _Float16 As[64][32];       // f16 tiles for WMMA
    __shared__ __align__(16) _Float16 Bs[64][32];

    const int t    = threadIdx.x;
    const int lane = t & 31;
    const int wave = t >> 5;
    const int wm   = wave >> 1;
    const int wn   = wave & 1;
    const int bm   = blockIdx.y * 64;
    const int bn   = blockIdx.x * 64;

    v8f acc0 = {}; v8f acc1 = {};
    const int nk = K >> 5;

    issue_tiles_async(A, W, K, N, bm, bn, 0, t, Af32[0], Bf32[0]);

    for (int i = 0; i < nk; ++i) {
        const int buf = i & 1;
        wait_async0();              // this wave's copies of tile i complete
        __syncthreads();            // -> whole tile visible to all waves

        if (i + 1 < nk)             // stream tile i+1 while we convert/compute
            issue_tiles_async(A, W, K, N, bm, bn, (i + 1) * 32, t,
                              Af32[buf ^ 1], Bf32[buf ^ 1]);

        // convert f32 staging -> f16 operand tiles (B transposed to [n][k])
        #pragma unroll
        for (int p = 0; p < 2; ++p) {
            int li = p * 256 + t;
            int r = li >> 3, kk = (li & 7) * 4;
            float4 v = *reinterpret_cast<const float4*>(&Af32[buf][r][kk]);
            H4 h; h.h[0] = (_Float16)v.x; h.h[1] = (_Float16)v.y;
                  h.h[2] = (_Float16)v.z; h.h[3] = (_Float16)v.w;
            *reinterpret_cast<uint2*>(&As[r][kk]) = h.u;
        }
        #pragma unroll
        for (int p = 0; p < 2; ++p) {
            int li = p * 256 + t;
            int kk = li >> 4, nn = (li & 15) * 4;
            float4 v = *reinterpret_cast<const float4*>(&Bf32[buf][kk][nn]);
            Bs[nn + 0][kk] = (_Float16)v.x;
            Bs[nn + 1][kk] = (_Float16)v.y;
            Bs[nn + 2][kk] = (_Float16)v.z;
            Bs[nn + 3][kk] = (_Float16)v.w;
        }
        __syncthreads();

        const int l15   = lane & 15;
        const int kbase = (lane >> 4) * 8;
        Frag16 fa, fb0, fb1;
        build_frag(fa,  &As[wm * 16 + l15][0], kbase);
        build_frag(fb0, &Bs[wn * 32 + l15][0], kbase);
        build_frag(fb1, &Bs[wn * 32 + 16 + l15][0], kbase);
        acc0 = __builtin_amdgcn_wmma_f32_16x16x32_f16(false, fa.v, false, fb0.v,
                                                      (short)0, acc0, false, false);
        acc1 = __builtin_amdgcn_wmma_f32_16x16x32_f16(false, fa.v, false, fb1.v,
                                                      (short)0, acc1, false, false);
        __syncthreads();
    }

    const int l15  = lane & 15;
    const int row0 = bm + wm * 16 + (lane >> 4) * 8;
    const int col0 = bn + wn * 32 + l15;
    const int col1 = col0 + 16;
    const float bc0 = bias[col0];
    const float bc1 = bias[col1];
    #pragma unroll
    for (int j = 0; j < 8; ++j) {
        int r = row0 + j;
        C[(long)r * N + col0] = acc0[j] + bc0;
        C[(long)r * N + col1] = acc1[j] + bc1;
    }
}

// ----------------------------- small helpers --------------------------------
__device__ inline void atomicMaxF(float* addr, float val) {
    if (val >= 0.0f)
        atomicMax(reinterpret_cast<int*>(addr), __float_as_int(val));
    else
        atomicMin(reinterpret_cast<unsigned int*>(addr), __float_as_uint(val));
}

__global__ void sum_kernel(const float* __restrict__ x, int n, float* __restrict__ out) {
    float s = 0.0f;
    for (int i = blockIdx.x * blockDim.x + threadIdx.x; i < n; i += gridDim.x * blockDim.x)
        s += x[i];
    for (int off = 16; off; off >>= 1) s += __shfl_down(s, off, 32);
    if ((threadIdx.x & 31) == 0) atomicAdd(out, s);
}

// k[h] = sum_c We[h*CH+c] * ae[h*CH+c]    (one 128-thread block, wave per head)
__global__ __launch_bounds__(128) void edge_coeff_kernel(
    const float* __restrict__ We, const float* __restrict__ ae, float* __restrict__ kout) {
    int h = threadIdx.x >> 5, lane = threadIdx.x & 31;
    float s = 0.0f;
    #pragma unroll
    for (int i = 0; i < 4; ++i) { int c = i * 32 + lane; s += We[h * CH + c] * ae[h * CH + c]; }
    for (int off = 16; off; off >>= 1) s += __shfl_down(s, off, 32);
    if (lane == 0) kout[h] = s;
}

// a_src[n,h], a_dst[n,h] dots; block per node, wave per head
__global__ __launch_bounds__(128) void node_coeff_kernel(
    const float* __restrict__ xw, const float* __restrict__ a_s, const float* __restrict__ a_d,
    float* __restrict__ asrc, float* __restrict__ adst, int n) {
    int node = blockIdx.x;
    if (node >= n) return;
    int h = threadIdx.x >> 5, lane = threadIdx.x & 31;
    const float* xr = xw + (long)node * HC + h * CH;
    float s1 = 0.0f, s2 = 0.0f;
    #pragma unroll
    for (int i = 0; i < 4; ++i) {
        int c = i * 32 + lane;
        float v = xr[c];
        s1 += v * a_s[h * CH + c];
        s2 += v * a_d[h * CH + c];
    }
    for (int off = 16; off; off >>= 1) {
        s1 += __shfl_down(s1, off, 32);
        s2 += __shfl_down(s2, off, 32);
    }
    if (lane == 0) { asrc[node * NHEADS + h] = s1; adst[node * NHEADS + h] = s2; }
}

// alpha = leaky_relu(a_src[s]+a_dst[d]+ea*k, 0.2); segment max via atomics
__global__ void alpha_max_kernel(
    const int* __restrict__ ei, const float* __restrict__ ea, const float* __restrict__ easum,
    const float* __restrict__ asrc, const float* __restrict__ adst, const float* __restrict__ kh,
    float* __restrict__ alpha, float* __restrict__ m, int E, int Etot) {
    int e = blockIdx.x * blockDim.x + threadIdx.x;
    if (e >= Etot) return;
    int s, d; float eav;
    if (e < E) { s = ei[e]; d = ei[E + e]; eav = ea[e]; }
    else       { s = d = e - E;            eav = easum[0] / (float)E; }
    #pragma unroll
    for (int h = 0; h < NHEADS; ++h) {
        float a = asrc[s * NHEADS + h] + adst[d * NHEADS + h] + eav * kh[h];
        a = (a > 0.0f) ? a : 0.2f * a;
        alpha[e * NHEADS + h] = a;
        atomicMaxF(&m[d * NHEADS + h], a);
    }
}

// ex = exp(alpha - m[d]); segment sum via atomics (in-place over alpha buf)
__global__ void exp_sum_kernel(
    const int* __restrict__ ei, float* __restrict__ alpha, const float* __restrict__ m,
    float* __restrict__ denom, int E, int Etot) {
    int e = blockIdx.x * blockDim.x + threadIdx.x;
    if (e >= Etot) return;
    int d = (e < E) ? ei[E + e] : (e - E);
    #pragma unroll
    for (int h = 0; h < NHEADS; ++h) {
        float ex = __expf(alpha[e * NHEADS + h] - m[d * NHEADS + h]);
        alpha[e * NHEADS + h] = ex;
        atomicAdd(&denom[d * NHEADS + h], ex);
    }
}

// att = ex/denom[d]; out[d,:] += xw[s,:]*att[head]; wave32 per edge, lanes cover 512 ch
__global__ __launch_bounds__(256) void att_agg_kernel(
    const int* __restrict__ ei, const float* __restrict__ ex, const float* __restrict__ denom,
    const float* __restrict__ xw, float* __restrict__ outagg, float* __restrict__ attsum,
    int E, int Etot) {
    int wave = threadIdx.x >> 5, lane = threadIdx.x & 31;
    long e = (long)blockIdx.x * 8 + wave;
    if (e >= Etot) return;
    int s, d;
    if (e < E) { s = ei[e]; d = ei[E + e]; } else { s = d = (int)e - E; }
    float att[NHEADS]; float asum = 0.0f;
    #pragma unroll
    for (int h = 0; h < NHEADS; ++h) {
        att[h] = ex[e * NHEADS + h] / (denom[d * NHEADS + h] + 1e-16f);
        asum += att[h];
    }
    if (lane == 0) atomicAdd(attsum, asum);
    const float* xs = xw + (long)s * HC;
    float* od = outagg + (long)d * HC;
    #pragma unroll
    for (int i = 0; i < 16; ++i) {
        int c = i * 32 + lane;
        atomicAdd(&od[c], xs[c] * att[c >> 7]);
    }
}

// per-column sum & sumsq of (x + bias); one block per column
__global__ __launch_bounds__(256) void colstats_kernel(
    const float* __restrict__ x, const float* __restrict__ bias,
    float* __restrict__ csum, float* __restrict__ csumsq, int rows, int cols) {
    int c = blockIdx.x;
    float b = bias[c];
    float s = 0.0f, sq = 0.0f;
    for (int r = threadIdx.x; r < rows; r += 256) {
        float v = x[(long)r * cols + c] + b;
        s += v; sq += v * v;
    }
    __shared__ float sh0[256], sh1[256];
    sh0[threadIdx.x] = s; sh1[threadIdx.x] = sq;
    __syncthreads();
    for (int off = 128; off; off >>= 1) {
        if (threadIdx.x < off) {
            sh0[threadIdx.x] += sh0[threadIdx.x + off];
            sh1[threadIdx.x] += sh1[threadIdx.x + off];
        }
        __syncthreads();
    }
    if (threadIdx.x == 0) { csum[c] = sh0[0]; csumsq[c] = sh1[0]; }
}

// GraphNorm (var from raw moments, bias folded) + ReLU
__global__ void norm_relu_kernel(
    const float* __restrict__ x, const float* __restrict__ bias,
    const float* __restrict__ csum, const float* __restrict__ csumsq,
    const float* __restrict__ g, const float* __restrict__ be, const float* __restrict__ ms,
    float* __restrict__ y, int rows, int cols) {
    long i = (long)blockIdx.x * blockDim.x + threadIdx.x;
    if (i >= (long)rows * cols) return;
    int c = (int)(i % cols);
    float inv = 1.0f / (float)rows;
    float mu  = csum[c] * inv;
    float msv = ms[c];
    float msq = csumsq[c] * inv;
    float var = msq - 2.0f * msv * mu * mu + msv * msv * mu * mu;
    float v = (x[i] + bias[c]) - msv * mu;
    float o = g[c] * v * rsqrtf(var + 1e-5f) + be[c];
    y[i] = (o > 0.0f) ? o : 0.0f;
}

// out[b] = dot(x[idx_drug[b]], Wout[0:512]) + dot(x[idx_cell[b]], Wout[512:1024]) + bout
__global__ __launch_bounds__(256) void head_kernel(
    const float* __restrict__ xn, const int* __restrict__ idxd, const int* __restrict__ idxc,
    const float* __restrict__ Wout, const float* __restrict__ bout, float* __restrict__ out, int B) {
    int wave = threadIdx.x >> 5, lane = threadIdx.x & 31;
    int b = blockIdx.x * 8 + wave;
    if (b >= B) return;
    const float* xd = xn + (long)idxd[b] * HC;
    const float* xc = xn + (long)idxc[b] * HC;
    float s = 0.0f;
    #pragma unroll
    for (int i = 0; i < 16; ++i) {
        int c = i * 32 + lane;
        s += xd[c] * Wout[c] + xc[c] * Wout[HC + c];
    }
    for (int off = 16; off; off >>= 1) s += __shfl_down(s, off, 32);
    if (lane == 0) out[b] = s + bout[0];
}

// all_att[s,d] = mean(att1)+mean(att2) on the edge set (matrix pre-zeroed)
__global__ void attmat_kernel(
    const int* __restrict__ ei, const float* __restrict__ as1, const float* __restrict__ as2,
    float* __restrict__ mat, int E, int Etot, int n) {
    int e = blockIdx.x * blockDim.x + threadIdx.x;
    if (e >= Etot) return;
    float v = (as1[0] + as2[0]) / ((float)Etot * (float)NHEADS);
    int s, d;
    if (e < E) { s = ei[e]; d = ei[E + e]; } else s = d = e - E;
    mat[(long)s * n + d] = v;
}

// ----------------------------- launcher -------------------------------------
extern "C" void kernel_launch(void* const* d_in, const int* in_sizes, int n_in,
                              void* d_out, int out_size, void* d_ws, size_t ws_size,
                              hipStream_t stream) {
    (void)n_in; (void)ws_size;
    const float* drug  = (const float*)d_in[0];
    const float* cell  = (const float*)d_in[1];
    const float* gene  = (const float*)d_in[2];
    const float* ea    = (const float*)d_in[3];
    const int*   ei    = (const int*)  d_in[4];
    const int*   idxd  = (const int*)  d_in[5];
    const int*   idxc  = (const int*)  d_in[6];
    const float* Wd    = (const float*)d_in[7];   const float* bd  = (const float*)d_in[8];
    const float* Wc    = (const float*)d_in[9];   const float* bc  = (const float*)d_in[10];
    const float* Wg    = (const float*)d_in[11];  const float* bg  = (const float*)d_in[12];
    const float* W1    = (const float*)d_in[13];  const float* b1  = (const float*)d_in[14];
    const float* as1w  = (const float*)d_in[15];  const float* ad1w = (const float*)d_in[16];
    const float* We1   = (const float*)d_in[17];  const float* ae1 = (const float*)d_in[18];
    const float* bias1 = (const float*)d_in[19];
    const float* g1    = (const float*)d_in[20];  const float* be1 = (const float*)d_in[21];
    const float* ms1   = (const float*)d_in[22];
    const float* W2    = (const float*)d_in[23];  const float* b2  = (const float*)d_in[24];
    const float* as2w  = (const float*)d_in[25];  const float* ad2w = (const float*)d_in[26];
    const float* We2   = (const float*)d_in[27];  const float* ae2 = (const float*)d_in[28];
    const float* bias2 = (const float*)d_in[29];
    const float* g2    = (const float*)d_in[30];  const float* be2 = (const float*)d_in[31];
    const float* ms2   = (const float*)d_in[32];
    const float* Wout  = (const float*)d_in[33];  const float* bo  = (const float*)d_in[34];

    const int E    = in_sizes[3];          // 160000
    const int Etot = E + NNODES;           // + self loops
    const int B    = in_sizes[5];          // 10000
    float* out = (float*)d_out;

    // workspace layout (floats)
    float* x0    = (float*)d_ws;                 // 8000*256
    float* xw    = x0    + (long)NNODES * 256;   // 8000*512
    float* agg   = xw    + (long)NNODES * HC;    // 8000*512
    float* xn    = agg   + (long)NNODES * HC;    // 8000*512
    float* asrc  = xn    + (long)NNODES * HC;    // 8000*4
    float* adst  = asrc  + NNODES * NHEADS;
    float* mbuf  = adst  + NNODES * NHEADS;
    float* denom = mbuf  + NNODES * NHEADS;
    float* exbuf = denom + NNODES * NHEADS;      // 168000*4
    float* csum  = exbuf + (long)Etot * NHEADS;  // 512
    float* csq   = csum  + HC;                   // 512
    float* easum = csq   + HC;                   // 1
    float* kh    = easum + 1;                    // 4
    float* att1s = kh    + NHEADS;               // 1
    float* att2s = att1s + 1;                    // 1

    // zero-init output (head region + 8000x8000 attention matrix) and scalars
    hipMemsetAsync(d_out, 0, (size_t)out_size * sizeof(float), stream);
    hipMemsetAsync(easum, 0, 8 * sizeof(float), stream);

    // mean edge attribute
    sum_kernel<<<512, 256, 0, stream>>>(ea, E, easum);

    // input projections -> x0 [8000,256]  (ragged shapes: general kernel)
    dim3 blk(256);
    gemm_f16_wmma<<<dim3(256 / 64, (1000 + 63) / 64), blk, 0, stream>>>(drug, Wd, bd, x0,                 1000, 1000, 256);
    gemm_f16_wmma<<<dim3(256 / 64, (1000 + 63) / 64), blk, 0, stream>>>(cell, Wc, bc, x0 + 1000L * 256,   1000, 1000, 256);
    gemm_f16_wmma<<<dim3(256 / 64, (6000 + 63) / 64), blk, 0, stream>>>(gene, Wg, bg, x0 + 2000L * 256,   6000, 6000, 256);

    const int eg = (Etot + 255) / 256;
    const int wg = (Etot + 7) / 8;

    // ---------------- layer 1 (aligned: async-pipelined GEMM) ----------------
    gemm_f16_wmma_async<<<dim3(HC / 64, NNODES / 64), blk, 0, stream>>>(x0, W1, b1, xw, NNODES, 256, HC);
    node_coeff_kernel<<<NNODES, 128, 0, stream>>>(xw, as1w, ad1w, asrc, adst, NNODES);
    edge_coeff_kernel<<<1, 128, 0, stream>>>(We1, ae1, kh);
    hipMemsetAsync(mbuf,  0xFF, NNODES * NHEADS * sizeof(float), stream);   // -NaN sentinel (always overwritten)
    hipMemsetAsync(denom, 0,    NNODES * NHEADS * sizeof(float), stream);
    hipMemsetAsync(agg,   0,    (size_t)NNODES * HC * sizeof(float), stream);
    alpha_max_kernel<<<eg, 256, 0, stream>>>(ei, ea, easum, asrc, adst, kh, exbuf, mbuf, E, Etot);
    exp_sum_kernel<<<eg, 256, 0, stream>>>(ei, exbuf, mbuf, denom, E, Etot);
    att_agg_kernel<<<wg, 256, 0, stream>>>(ei, exbuf, denom, xw, agg, att1s, E, Etot);
    colstats_kernel<<<HC, 256, 0, stream>>>(agg, bias1, csum, csq, NNODES, HC);
    norm_relu_kernel<<<(NNODES * HC + 255) / 256, 256, 0, stream>>>(agg, bias1, csum, csq, g1, be1, ms1, xn, NNODES, HC);

    // ---------------- layer 2 (aligned: async-pipelined GEMM) ----------------
    gemm_f16_wmma_async<<<dim3(HC / 64, NNODES / 64), blk, 0, stream>>>(xn, W2, b2, xw, NNODES, HC, HC);
    node_coeff_kernel<<<NNODES, 128, 0, stream>>>(xw, as2w, ad2w, asrc, adst, NNODES);
    edge_coeff_kernel<<<1, 128, 0, stream>>>(We2, ae2, kh);
    hipMemsetAsync(mbuf,  0xFF, NNODES * NHEADS * sizeof(float), stream);
    hipMemsetAsync(denom, 0,    NNODES * NHEADS * sizeof(float), stream);
    hipMemsetAsync(agg,   0,    (size_t)NNODES * HC * sizeof(float), stream);
    alpha_max_kernel<<<eg, 256, 0, stream>>>(ei, ea, easum, asrc, adst, kh, exbuf, mbuf, E, Etot);
    exp_sum_kernel<<<eg, 256, 0, stream>>>(ei, exbuf, mbuf, denom, E, Etot);
    att_agg_kernel<<<wg, 256, 0, stream>>>(ei, exbuf, denom, xw, agg, att2s, E, Etot);
    colstats_kernel<<<HC, 256, 0, stream>>>(agg, bias2, csum, csq, NNODES, HC);
    norm_relu_kernel<<<(NNODES * HC + 255) / 256, 256, 0, stream>>>(agg, bias2, csum, csq, g2, be2, ms2, xn, NNODES, HC);

    // ---------------- outputs ----------------
    head_kernel<<<(B + 7) / 8, 256, 0, stream>>>(xn, idxd, idxc, Wout, bo, out, B);
    attmat_kernel<<<eg, 256, 0, stream>>>(ei, att1s, att2s, out + B, E, Etot, NNODES);
}
